// MoE_22385369546978
// MI455X (gfx1250) — matrix operand order repaired and checked by reading, not compiled, not run on hardware
//
#include <hip/hip_runtime.h>

// ---------------- problem constants ----------------
#define Bsz 16384
#define Dd  1024
#define Ee  8
#define Hh  1024
#define BM  64          // rows of x per workgroup
#define XP  1040        // LDS pitch for x tile
#define HP  264         // LDS pitch for h chunk
#define WP  40          // LDS pitch for weight slab rows (32 K + 8 pad) == TDM pad output

typedef __attribute__((ext_vector_type(16))) __bf16 v16bf;
typedef __attribute__((ext_vector_type(8)))  float  v8f;
typedef __attribute__((ext_vector_type(4)))  unsigned int u32x4;
typedef __attribute__((ext_vector_type(8)))  int i32x8;
typedef __attribute__((ext_vector_type(4)))  int i32x4;

union FragU { uint4 u[2]; v16bf v; };

static __device__ __forceinline__ unsigned short f2bf(float f) {
  unsigned u = __float_as_uint(f);
  return (unsigned short)((u + 0x7FFFu + ((u >> 16) & 1u)) >> 16);
}

// A-fragment (16x32 bf16): lane<16 -> m=lane, K {0..7, 16..23}; lane>=16 -> K {8..15, 24..31}
static __device__ __forceinline__ v16bf ldA(const unsigned short* base, int pitch,
                                            int row0, int k0, int lane) {
  int m  = row0 + (lane & 15);
  int ko = (lane >> 4) * 8;
  FragU f;
  f.u[0] = *(const uint4*)(base + m * pitch + k0 + ko);
  f.u[1] = *(const uint4*)(base + m * pitch + k0 + 16 + ko);
  return f.v;
}

// B-fragment (32x16 bf16) from K-contiguous slab: lane<16 -> n=lane, K {0..7, 16..23}; lane>=16 -> K {8..15, 24..31}
static __device__ __forceinline__ v16bf ldB(const unsigned short* base, int n0, int lane) {
  int n  = n0 + (lane & 15);
  int ko = (lane >> 4) * 8;
  FragU f;
  f.u[0] = *(const uint4*)(base + n * WP + ko);
  f.u[1] = *(const uint4*)(base + n * WP + ko + 16);
  return f.v;
}

#define WMMA_BF16(A, Bm, C) \
  __builtin_amdgcn_wmma_f32_16x16x32_bf16(false, (A), false, (Bm), (short)0, (C), false, false)

// ---- Tensor Data Mover: 2-D tile [rows x 32] bf16, row stride 1024 elems, LDS-padded to WP ----
static __device__ __forceinline__ void tdm_issue(unsigned lds_byte_off,
                                                 const unsigned short* gaddr, int rows) {
  unsigned long long va = (unsigned long long)(size_t)gaddr;
  u32x4 g0;
  g0[0] = 1u;                                             // count=1, user descriptor
  g0[1] = lds_byte_off;                                   // lds_addr
  g0[2] = (unsigned)(va & 0xFFFFFFFFu);                   // global_addr[31:0]
  g0[3] = (unsigned)((va >> 32) & 0x01FFFFFFu) | (2u << 30);  // global_addr[56:32] | type=2
  i32x8 g1;
  // data_size=2B(1), pad_enable, pad_interval=16dw (64B row), pad_amount=4dw (8 bf16)
  g1[0] = (1 << 16) | (1 << 20) | (3 << 22) | (3 << 25);
  g1[1] = (int)(1024u << 16);       // tensor_dim0[15:0]=1024 in bits 63:48
  g1[2] = (int)(1024u << 16);       // tensor_dim0 hi=0 | tensor_dim1[15:0]=1024
  g1[3] = (int)(32u << 16);         // tensor_dim1 hi=0 | tile_dim0=32
  g1[4] = rows;                     // tile_dim1=rows | tile_dim2=0
  g1[5] = 1024;                     // tensor_dim0_stride[31:0]
  g1[6] = 0;
  g1[7] = 0;
  i32x4 z4 = {0, 0, 0, 0};
#if __clang_major__ >= 23
  i32x8 z8 = {0, 0, 0, 0, 0, 0, 0, 0};
  __builtin_amdgcn_tensor_load_to_lds(g0, g1, z4, z4, z8, 0);
#else
  __builtin_amdgcn_tensor_load_to_lds(g0, g1, z4, z4, 0);
#endif
}

static __device__ __forceinline__ unsigned lds_off(const void* p) {
  return (unsigned)(size_t)p;  // generic ptr low 32 bits = LDS byte offset (aperture layout)
}

// ---------------- kernel 1a: fp32 -> bf16 cast (x, vectorized) ----------------
__global__ void cast_bf16_kernel(const float* __restrict__ src,
                                 unsigned short* __restrict__ dst, int n4) {
  int i = blockIdx.x * blockDim.x + threadIdx.x;
  if (i < n4) {
    float4 f = ((const float4*)src)[i];
    uint2 o;
    o.x = (unsigned)f2bf(f.x) | ((unsigned)f2bf(f.y) << 16);
    o.y = (unsigned)f2bf(f.z) | ((unsigned)f2bf(f.w) << 16);
    ((uint2*)dst)[i] = o;
  }
}

// ---------------- kernel 1b: fused cast + transpose for weights ----------------
// src: [E][R][C] fp32  ->  dst: [E][C][R] bf16   (32x32 LDS tiles, coalesced both sides)
__global__ __launch_bounds__(256) void cast_transpose_kernel(const float* __restrict__ src,
                                                             unsigned short* __restrict__ dst,
                                                             int R, int C) {
  __shared__ unsigned short t[32][33];
  int e = blockIdx.z;
  src += (size_t)e * R * C;
  dst += (size_t)e * R * C;
  int c0 = blockIdx.x * 32, r0 = blockIdx.y * 32;
  int tx = threadIdx.x & 31, ty = threadIdx.x >> 5;
#pragma unroll
  for (int j = 0; j < 4; ++j)
    t[ty + j * 8][tx] = f2bf(src[(size_t)(r0 + ty + j * 8) * C + c0 + tx]);
  __syncthreads();
#pragma unroll
  for (int j = 0; j < 4; ++j)
    dst[(size_t)(c0 + ty + j * 8) * R + r0 + tx] = t[tx][ty + j * 8];
}

// ---------------- kernel 2: router softmax gates ----------------
__global__ __launch_bounds__(256) void router_kernel(const float* __restrict__ x,
                                                     const float* __restrict__ Wr,
                                                     const float* __restrict__ br,
                                                     float* __restrict__ gates) {
  int lane = threadIdx.x & 31;
  int wid  = threadIdx.x >> 5;
  int b    = blockIdx.x * 8 + wid;
  float acc[Ee];
#pragma unroll
  for (int e = 0; e < Ee; ++e) acc[e] = 0.f;
  for (int k = lane; k < Dd; k += 32) {
    float xv = x[(size_t)b * Dd + k];
#pragma unroll
    for (int e = 0; e < Ee; ++e) acc[e] += xv * Wr[k * Ee + e];
  }
#pragma unroll
  for (int off = 16; off > 0; off >>= 1)
#pragma unroll
    for (int e = 0; e < Ee; ++e) acc[e] += __shfl_xor(acc[e], off, 32);
  if (lane == 0) {
    float m = -3.4e38f;
#pragma unroll
    for (int e = 0; e < Ee; ++e) { acc[e] += br[e]; m = fmaxf(m, acc[e]); }
    float s = 0.f;
#pragma unroll
    for (int e = 0; e < Ee; ++e) { acc[e] = __expf(acc[e] - m); s += acc[e]; }
    float inv = 1.f / s;
#pragma unroll
    for (int e = 0; e < Ee; ++e) gates[(size_t)b * Ee + e] = acc[e] * inv;
  }
}

// ---------------- kernel 3: fused MoE with TDM double-buffered weight slabs ----------------
__global__ __launch_bounds__(256) void moe_kernel(const unsigned short* __restrict__ xb,
                                                  const unsigned short* __restrict__ w1t,
                                                  const unsigned short* __restrict__ w2t,
                                                  const float* __restrict__ gates,
                                                  const float* __restrict__ b1,
                                                  const float* __restrict__ b2,
                                                  const float* __restrict__ x,
                                                  float* __restrict__ out) {
  __shared__ __align__(16) unsigned short lds_x[BM * XP];       // 133,120 B
  __shared__ __align__(16) unsigned short lds_h[BM * HP];       //  33,792 B
  __shared__ __align__(16) unsigned short lds_wt[2][512 * WP];  //  81,920 B (double buffer)
  __shared__ float lds_g[BM * Ee];                              //   2,048 B

  const int tid  = threadIdx.x;
  const int lane = tid & 31;
  const int wid  = tid >> 5;
  const int wm   = wid >> 2;   // 0..1
  const int wn   = wid & 3;    // 0..3
  const int bm0  = blockIdx.x * BM;
  const unsigned wt_off0 = lds_off(&lds_wt[0][0]);
  const unsigned wt_off1 = lds_off(&lds_wt[1][0]);

  // stage x tile [64 x 1024] bf16 and gates into LDS
  for (int i = tid; i < BM * (Dd / 8); i += 256) {
    int r = i >> 7, c = i & 127;
    *(uint4*)(lds_x + r * XP + c * 8) = *(const uint4*)(xb + (size_t)(bm0 + r) * Dd + c * 8);
  }
  for (int i = tid; i < BM * Ee; i += 256) lds_g[i] = gates[(size_t)bm0 * Ee + i];

  const v8f z8 = {0.f, 0.f, 0.f, 0.f, 0.f, 0.f, 0.f, 0.f};
  v8f oacc[2][16];
#pragma unroll
  for (int a = 0; a < 2; ++a)
#pragma unroll
    for (int c = 0; c < 16; ++c) oacc[a][c] = z8;

  __syncthreads();  // x tile visible before first use

  for (int e = 0; e < Ee; ++e) {
    const unsigned short* w1e = w1t + (size_t)e * Dd * Hh;  // [H][D] bf16
    const unsigned short* w2e = w2t + (size_t)e * Hh * Dd;  // [D][H] bf16
    for (int nc = 0; nc < 4; ++nc) {
      v8f hacc[2][4];
#pragma unroll
      for (int a = 0; a < 2; ++a)
#pragma unroll
        for (int c = 0; c < 4; ++c) hacc[a][c] = z8;

      // ---- GEMM1: h_chunk = x @ W1[e][:, nc*256:+256]  (slabs: [256 n][32 k]) ----
      const unsigned short* s1base = w1e + (size_t)(nc * 256) * Dd;
      if (wid == 0) tdm_issue(wt_off0, s1base, 256);
      for (int kk = 0; kk < Dd / 32; ++kk) {
        if (wid == 0) {
          if (kk < 31) {
            tdm_issue((kk & 1) ? wt_off0 : wt_off1, s1base + (size_t)(kk + 1) * 32, 256);
            __builtin_amdgcn_s_wait_tensorcnt(1);
          } else {
            __builtin_amdgcn_s_wait_tensorcnt(0);
          }
        }
        __syncthreads();
        const unsigned short* wb = &lds_wt[kk & 1][0];
        v16bf a0 = ldA(lds_x, XP, wm * 32,      kk * 32, lane);
        v16bf a1 = ldA(lds_x, XP, wm * 32 + 16, kk * 32, lane);
        // software-pipelined B fragments: load nt+1 before consuming nt
        v16bf bcur = ldB(wb, wn * 64, lane);
#pragma unroll
        for (int nt = 0; nt < 4; ++nt) {
          v16bf bnext = bcur;
          if (nt < 3) bnext = ldB(wb, wn * 64 + (nt + 1) * 16, lane);
          hacc[0][nt] = WMMA_BF16(a0, bcur, hacc[0][nt]);
          hacc[1][nt] = WMMA_BF16(a1, bcur, hacc[1][nt]);
          bcur = bnext;
        }
        __syncthreads();
      }

      // ---- epilogue1: bias + swish + gate -> bf16 h chunk in LDS ----
#pragma unroll
      for (int mt = 0; mt < 2; ++mt)
#pragma unroll
        for (int nt = 0; nt < 4; ++nt) {
          int nl = wn * 64 + nt * 16 + (lane & 15);
          float bias = b1[e * Hh + nc * 256 + nl];
#pragma unroll
          for (int t = 0; t < 8; ++t) {
            int ml = wm * 32 + mt * 16 + (lane >> 4) * 8 + t;
            float z  = hacc[mt][nt][t] + bias;
            float sw = z / (1.f + __expf(-z));          // swish
            lds_h[ml * HP + nl] = f2bf(lds_g[ml * Ee + e] * sw);
          }
        }
      __syncthreads();

      // ---- GEMM2 partial: out += h_chunk @ W2[e][nc*256:+256, :]  (slabs: [512 n][32 k]) ----
      if (wid == 0) tdm_issue(wt_off0, w2e + (size_t)(nc * 256), 512);
      for (int k2 = 0; k2 < 8; ++k2) {
        // A fragments are invariant across the two n-halves of this k2 step
        v16bf a0 = ldA(lds_h, HP, wm * 32,      k2 * 32, lane);
        v16bf a1 = ldA(lds_h, HP, wm * 32 + 16, k2 * 32, lane);
#pragma unroll
        for (int half = 0; half < 2; ++half) {
          if (wid == 0) {
            if (k2 * 2 + half < 15) {
              int ns  = k2 * 2 + half + 1;
              int nh  = ns & 1;
              int nk2 = ns >> 1;
              tdm_issue((ns & 1) ? wt_off1 : wt_off0,
                        w2e + (size_t)nh * 512 * Dd + (size_t)(nc * 256 + nk2 * 32), 512);
              __builtin_amdgcn_s_wait_tensorcnt(1);
            } else {
              __builtin_amdgcn_s_wait_tensorcnt(0);
            }
          }
          __syncthreads();
          const unsigned short* wb = &lds_wt[half][0];   // s = 2*k2+half -> buf = half
          // software-pipelined B fragments
          v16bf bcur = ldB(wb, wn * 128, lane);
#pragma unroll
          for (int nt = 0; nt < 8; ++nt) {
            v16bf bnext = bcur;
            if (nt < 7) bnext = ldB(wb, wn * 128 + (nt + 1) * 16, lane);
            int j = half * 8 + nt;
            oacc[0][j] = WMMA_BF16(a0, bcur, oacc[0][j]);
            oacc[1][j] = WMMA_BF16(a1, bcur, oacc[1][j]);
            bcur = bnext;
          }
          __syncthreads();
        }
      }
    }
  }

  // ---- epilogue2: + sum_e g*b2[e] + residual x ----
#pragma unroll
  for (int j = 0; j < 16; ++j) {
    int n = (j >> 3) * 512 + wn * 128 + (j & 7) * 16 + (lane & 15);
    float bb[Ee];
#pragma unroll
    for (int e = 0; e < Ee; ++e) bb[e] = b2[e * Dd + n];
#pragma unroll
    for (int mt = 0; mt < 2; ++mt)
#pragma unroll
      for (int t = 0; t < 8; ++t) {
        int ml = wm * 32 + mt * 16 + (lane >> 4) * 8 + t;
        size_t g = (size_t)(bm0 + ml) * Dd + n;
        float v = oacc[mt][j][t];
#pragma unroll
        for (int e = 0; e < Ee; ++e) v += lds_g[ml * Ee + e] * bb[e];
        out[g] = v + x[g];
      }
  }
}

// ---------------- host launcher ----------------
extern "C" void kernel_launch(void* const* d_in, const int* in_sizes, int n_in,
                              void* d_out, int out_size, void* d_ws, size_t ws_size,
                              hipStream_t stream) {
  const float* x  = (const float*)d_in[0];
  const float* W1 = (const float*)d_in[1];
  const float* b1 = (const float*)d_in[2];
  const float* W2 = (const float*)d_in[3];
  const float* b2 = (const float*)d_in[4];
  const float* Wr = (const float*)d_in[5];
  const float* br = (const float*)d_in[6];
  float* out = (float*)d_out;

  // workspace: xb (32MB) | w1t (16MB, [E][H][D]) | w2t (16MB, [E][D][H]) | gates (512KB)
  unsigned short* xb  = (unsigned short*)d_ws;
  unsigned short* w1t = xb  + (size_t)Bsz * Dd;
  unsigned short* w2t = w1t + (size_t)Ee * Dd * Hh;
  float*          gts = (float*)(w2t + (size_t)Ee * Hh * Dd);

  int n4x = Bsz * Dd / 4;
  cast_bf16_kernel<<<(n4x + 255) / 256, 256, 0, stream>>>(x, xb, n4x);

  dim3 tgrid(Hh / 32, Dd / 32, Ee);
  cast_transpose_kernel<<<tgrid, 256, 0, stream>>>(W1, w1t, Dd, Hh);  // -> [E][H][D]
  cast_transpose_kernel<<<tgrid, 256, 0, stream>>>(W2, w2t, Hh, Dd);  // -> [E][D][H]

  router_kernel<<<Bsz / 8, 256, 0, stream>>>(x, Wr, br, gts);

  moe_kernel<<<Bsz / BM, 256, 0, stream>>>(xb, w1t, w2t, gts, b1, b2, x, out);
}
